// Attention_20993800143414
// MI455X (gfx1250) — matrix operand order repaired and checked by reading, not compiled, not run on hardware
//
#include <hip/hip_runtime.h>
#include <hip/hip_bf16.h>

typedef __attribute__((ext_vector_type(16))) _Float16 v16h;
typedef __attribute__((ext_vector_type(8)))  _Float16 v8h;
typedef __attribute__((ext_vector_type(8)))  float    v8f;

#define NB   2
#define SEQ  2048
#define CH   256
#define NH   8
#define HD   32
#define KB   64                       // keys per flash-attention block
#define QSCALE 0.17677669529663687f   // 32^-0.5

#define WMMA_F16(A, B, Cacc) \
  __builtin_amdgcn_wmma_f32_16x16x32_f16(false, (A), false, (B), (short)0, (Cacc), false, false)

// ---- operand builders ---------------------------------------------------
// A 16x32 f16 layout: lane L holds row M=L%16; koff = 8*(L/16);
// elements 0..15 = K[2*koff..2*koff+15] (= K[16*(L/16) + 0..15]).
static __device__ __forceinline__ v16h cvt_a_f32(const float* row, int koff) {
  v16h r;
#pragma unroll
  for (int i = 0; i < 8; ++i) {
    r[i]     = (_Float16)row[2 * koff + i];
    r[i + 8] = (_Float16)row[2 * koff + 8 + i];
  }
  return r;
}
static __device__ __forceinline__ v16h load_a_f16(const _Float16* row, int koff) {
  v8h lo = *(const v8h*)(row + 2 * koff);
  v8h hi = *(const v8h*)(row + 2 * koff + 8);
  v16h r;
#pragma unroll
  for (int i = 0; i < 8; ++i) { r[i] = lo[i]; r[i + 8] = hi[i]; }
  return r;
}
// B 32x16 f16 layout: lane L holds col N=L%16; elements 0..15 = K[16*(L/16) + 0..15]
static __device__ __forceinline__ v16h cvt_b_f32(const float* p) {
  v16h r;
#pragma unroll
  for (int i = 0; i < 16; ++i) r[i] = (_Float16)p[i];
  return r;
}

// ---- kernel 1: q/k/v/gate projections -----------------------------------
// mat 0: qh[b][h][s][d] = f16( (q_x @ Wq.T) * QSCALE )
// mat 1: kh[b][h][s][d] = f16(  kv_x @ Wk.T )
// mat 2: vt[b][ch][s]   = f16( (kv_x @ Wv.T).T )   (computed as Wv @ kv_x^T)
// mat 3: g[r][ch]       = sigmoid(q_x @ Wg.T + b_g + gating_bias)   (f32)
__global__ __launch_bounds__(256) void proj_kernel(
    const float* __restrict__ qx, const float* __restrict__ kvx,
    const float* __restrict__ Wq, const float* __restrict__ Wk,
    const float* __restrict__ Wv, const float* __restrict__ Wg,
    const float* __restrict__ bg, const float* __restrict__ gbias,
    _Float16* __restrict__ qh, _Float16* __restrict__ kh,
    _Float16* __restrict__ vt, float* __restrict__ gout) {
  const int warp = threadIdx.x >> 5, lane = threadIdx.x & 31;
  const int n0 = lane & 15, lh = lane >> 4, koff = lh * 8;
  const int gw  = blockIdx.x * 8 + warp;        // 0..4095
  const int mat = gw >> 10;                     // 0..3
  const int rem = gw & 1023;
  int mtile, ng;
  if (mat == 2) { mtile = rem >> 6; ng = rem & 63; }   // M=256 (ch), N=4096 (r)
  else          { mtile = rem >> 2; ng = rem & 3;  }   // M=4096 (r), N=256 (ch)
  const int mbase = mtile * 16, nbase = ng * 64;

  const float* Asrc = (mat == 1) ? kvx : (mat == 2 ? Wv : qx);
  const float* Bsrc = (mat == 0) ? Wq : (mat == 1 ? Wk : (mat == 2 ? kvx : Wg));
  const float* arow = Asrc + (size_t)(mbase + n0) * CH;

  v8f acc[4] = {};
  for (int ko = 0; ko < CH; ko += 32) {
    v16h av = cvt_a_f32(arow + ko, koff);
#pragma unroll
    for (int t = 0; t < 4; ++t) {
      const float* brow = Bsrc + (size_t)(nbase + t * 16 + n0) * CH + ko + lh * 16;
      v16h bv = cvt_b_f32(brow);
      acc[t] = WMMA_F16(av, bv, acc[t]);
    }
  }

#pragma unroll
  for (int t = 0; t < 4; ++t) {
#pragma unroll
    for (int j = 0; j < 8; ++j) {
      float c = acc[t][j];
      if (mat == 2) {
        int chrow = mbase + j + 8 * lh;
        int r = nbase + t * 16 + n0;
        int b = r >> 11, s = r & (SEQ - 1);
        vt[((size_t)(b * CH + chrow)) * SEQ + s] = (_Float16)c;
      } else {
        int r  = mbase + j + 8 * lh;
        int ch = nbase + t * 16 + n0;
        if (mat == 3) {
          float v = c + bg[ch] + gbias[ch];
          gout[(size_t)r * CH + ch] = 1.0f / (1.0f + __expf(-v));
        } else {
          int b = r >> 11, s = r & (SEQ - 1), h = ch >> 5, d = ch & 31;
          size_t idx = (((size_t)(b * NH + h)) * SEQ + s) * HD + d;
          if (mat == 0) qh[idx] = (_Float16)(c * QSCALE);
          else          kh[idx] = (_Float16)c;
        }
      }
    }
  }
}

// ---- kernel 2: flash attention ------------------------------------------
// Block = 8 waves, one (b,h), 128 q rows. Per 64-key step the block DMAs the
// shared K (64x32 f16) and V^T (32x64 f16) tiles into double-buffered LDS via
// gfx1250 async global->LDS loads (ASYNCcnt), overlapping the next fetch with
// the current step's WMMAs. Bias stream (256 MB total) stays direct-to-VGPR.
__global__ __launch_bounds__(256) void attn_kernel(
    const float* __restrict__ bias, const _Float16* __restrict__ qh,
    const _Float16* __restrict__ kh, const _Float16* __restrict__ vt,
    _Float16* __restrict__ oh) {
  __shared__ __align__(64) _Float16 kbuf[2][KB][HD];       // 2 x 4 KB [key][d]
  __shared__ __align__(64) _Float16 vbuf[2][HD][KB];       // 2 x 4 KB [d][key]
  __shared__ __align__(64) _Float16 pbuf[8][16 * KB];      // 16 KB P re-layout
  const int tid = threadIdx.x;
  const int warp = tid >> 5, lane = tid & 31;
  const int n0 = lane & 15, lh = lane >> 4, koff = lh * 8;
  const int bh = blockIdx.x >> 4;        // b*NH + h
  const int qc = blockIdx.x & 15;
  const int b = bh >> 3, h = bh & 7;
  const int qbase = qc * 128 + warp * 16;

  const _Float16* kbp = kh + (size_t)bh * SEQ * HD;
  const _Float16* vbp = vt + (size_t)(b * CH + h * HD) * SEQ;

  // cooperative async-DMA mapping: 256 threads x (one 16B K chunk + one 16B V chunk)
  const int krow = tid >> 2, kc = tid & 3;   // K: 64 rows x 4 chunks of 8 halfs
  const int vrow = tid >> 3, vc = tid & 7;   // V: 32 rows x 8 chunks of 8 halfs

  auto issue_async = [&](int kb, int bufi) {
    unsigned kl = (unsigned)(uintptr_t)&kbuf[bufi][krow][kc * 8];
    unsigned long long kg =
        (unsigned long long)(uintptr_t)(kbp + (size_t)(kb + krow) * HD + kc * 8);
    asm volatile("global_load_async_to_lds_b128 %0, %1, off"
                 :: "v"(kl), "v"(kg) : "memory");
    unsigned vl = (unsigned)(uintptr_t)&vbuf[bufi][vrow][vc * 8];
    unsigned long long vg =
        (unsigned long long)(uintptr_t)(vbp + (size_t)vrow * SEQ + kb + vc * 8);
    asm volatile("global_load_async_to_lds_b128 %0, %1, off"
                 :: "v"(vl), "v"(vg) : "memory");
  };

  issue_async(0, 0);  // prime the pipeline

  const _Float16* qrow = qh + (((size_t)bh) * SEQ + qbase + n0) * HD;
  const v16h qa = load_a_f16(qrow, koff);

  v8f o0 = {}, o1 = {};
  float m[8], l[8];
#pragma unroll
  for (int j = 0; j < 8; ++j) { m[j] = -3.0e38f; l[j] = 0.0f; }

  const float* biasq = bias + (((size_t)bh) * SEQ + qbase) * SEQ;
  _Float16* pb = pbuf[warp];

  const int NIT = SEQ / KB;  // 32
  for (int i = 0; i < NIT; ++i) {
    const int kb = i * KB;
    const int cur = i & 1;
    // own async data for step i has landed:
    asm volatile("s_wait_asynccnt 0" ::: "memory");
    // prefetch step i+1 into the other buffer (its readers finished at the
    // end-of-step barrier of step i-1):
    if (i + 1 < NIT) issue_async(kb + KB, cur ^ 1);
    __syncthreads();  // all threads' step-i tiles visible

    // scores: 4 WMMAs (Q 16x32 x K^T 32x16 per 16-key tile)
    v8f s[4];
#pragma unroll
    for (int T = 0; T < 4; ++T) {
      v16h kt = *(const v16h*)(&kbuf[cur][T * 16 + n0][lh * 16]);
      v8f z = {};
      s[T] = WMMA_F16(qa, kt, z);
    }

    // additive bias (dominant HBM traffic; coalesced + next-block prefetch)
#pragma unroll
    for (int j = 0; j < 8; ++j) {
      const float* bp = biasq + (size_t)(j + 8 * lh) * SEQ + kb + n0;
      s[0][j] += bp[0];
      s[1][j] += bp[16];
      s[2][j] += bp[32];
      s[3][j] += bp[48];
      if (kb + KB < SEQ) __builtin_prefetch(bp + KB, 0, 0);  // global_prefetch_b8
    }

    // online softmax; one 4-step xor-reduction pair per row per 64 keys
#pragma unroll
    for (int j = 0; j < 8; ++j) {
      float mt = fmaxf(fmaxf(s[0][j], s[1][j]), fmaxf(s[2][j], s[3][j]));
      mt = fmaxf(mt, __shfl_xor(mt, 8));
      mt = fmaxf(mt, __shfl_xor(mt, 4));
      mt = fmaxf(mt, __shfl_xor(mt, 2));
      mt = fmaxf(mt, __shfl_xor(mt, 1));
      float mn = fmaxf(m[j], mt);
      float alpha = __expf(m[j] - mn);
      float p0 = __expf(s[0][j] - mn);
      float p1 = __expf(s[1][j] - mn);
      float p2 = __expf(s[2][j] - mn);
      float p3 = __expf(s[3][j] - mn);
      float rs = (p0 + p1) + (p2 + p3);
      rs += __shfl_xor(rs, 8);
      rs += __shfl_xor(rs, 4);
      rs += __shfl_xor(rs, 2);
      rs += __shfl_xor(rs, 1);
      l[j] = l[j] * alpha + rs;
      m[j] = mn;
      o0[j] *= alpha;
      o1[j] *= alpha;
      const int rr = (j + 8 * lh) * KB;   // C-layout -> row-major [16 m][64 k]
      pb[rr + n0]      = (_Float16)p0;
      pb[rr + 16 + n0] = (_Float16)p1;
      pb[rr + 32 + n0] = (_Float16)p2;
      pb[rr + 48 + n0] = (_Float16)p3;
    }
    __syncthreads();  // pbuf ready

    const _Float16* prow = pb + n0 * KB;
    v16h pa0 = load_a_f16(prow, koff);        // k = 0..31
    v16h pa1 = load_a_f16(prow + 32, koff);   // k = 32..63
    v16h vb00 = *(const v16h*)(&vbuf[cur][n0][lh * 16]);            // d 0..15, k 0..31
    v16h vb10 = *(const v16h*)(&vbuf[cur][n0][32 + lh * 16]);       // d 0..15, k 32..63
    v16h vb01 = *(const v16h*)(&vbuf[cur][16 + n0][lh * 16]);       // d 16..31, k 0..31
    v16h vb11 = *(const v16h*)(&vbuf[cur][16 + n0][32 + lh * 16]);  // d 16..31, k 32..63
    o0 = WMMA_F16(pa0, vb00, o0);
    o0 = WMMA_F16(pa1, vb10, o0);
    o1 = WMMA_F16(pa0, vb01, o1);
    o1 = WMMA_F16(pa1, vb11, o1);
    __syncthreads();  // step's LDS reads done -> buffer may be overwritten
  }

#pragma unroll
  for (int j = 0; j < 8; ++j) {
    float inv = 1.0f / l[j];
    size_t r = (size_t)b * SEQ + qbase + j + 8 * lh;
    oh[r * CH + h * HD + n0]      = (_Float16)(o0[j] * inv);
    oh[r * CH + h * HD + 16 + n0] = (_Float16)(o1[j] * inv);
  }
}

// ---- kernel 3: output projection + bias + sigmoid gate ------------------
__global__ __launch_bounds__(256) void outproj_kernel(
    const _Float16* __restrict__ oh, const float* __restrict__ Wout,
    const float* __restrict__ bout, const float* __restrict__ g,
    float* __restrict__ out) {
  const int warp = threadIdx.x >> 5, lane = threadIdx.x & 31;
  const int n0 = lane & 15, lh = lane >> 4, koff = lh * 8;
  const int gw = blockIdx.x * 8 + warp;  // 0..1023
  const int mbase = (gw >> 2) * 16, nbase = (gw & 3) * 64;

  const _Float16* arow = oh + (size_t)(mbase + n0) * CH;
  v8f acc[4] = {};
  for (int ko = 0; ko < CH; ko += 32) {
    v16h av = load_a_f16(arow + ko, koff);
#pragma unroll
    for (int t = 0; t < 4; ++t) {
      const float* brow = Wout + (size_t)(nbase + t * 16 + n0) * CH + ko + lh * 16;
      v16h bv = cvt_b_f32(brow);
      acc[t] = WMMA_F16(av, bv, acc[t]);
    }
  }
#pragma unroll
  for (int t = 0; t < 4; ++t) {
    int ch = nbase + t * 16 + n0;
    float bo = bout[ch];
#pragma unroll
    for (int j = 0; j < 8; ++j) {
      size_t r = mbase + j + 8 * lh;
      out[r * CH + ch] = (acc[t][j] + bo) * g[r * CH + ch];
    }
  }
}

// ---- launch -------------------------------------------------------------
extern "C" void kernel_launch(void* const* d_in, const int* in_sizes, int n_in,
                              void* d_out, int out_size, void* d_ws, size_t ws_size,
                              hipStream_t stream) {
  const float* qx    = (const float*)d_in[0];
  const float* kvx   = (const float*)d_in[1];
  const float* bias  = (const float*)d_in[2];
  const float* Wq    = (const float*)d_in[3];
  const float* Wk    = (const float*)d_in[4];
  const float* Wv    = (const float*)d_in[5];
  const float* Wout  = (const float*)d_in[6];
  const float* bout  = (const float*)d_in[7];
  const float* Wg    = (const float*)d_in[8];
  const float* bg    = (const float*)d_in[9];
  const float* gbias = (const float*)d_in[10];
  float* out = (float*)d_out;

  char* ws = (char*)d_ws;
  _Float16* qh = (_Float16*)(ws + (0u << 20));   // 2 MiB
  _Float16* kh = (_Float16*)(ws + (2u << 20));   // 2 MiB
  _Float16* vt = (_Float16*)(ws + (4u << 20));   // 2 MiB
  _Float16* oh = (_Float16*)(ws + (6u << 20));   // 2 MiB
  float*    gb = (float*)   (ws + (8u << 20));   // 4 MiB

  proj_kernel<<<512, 256, 0, stream>>>(qx, kvx, Wq, Wk, Wv, Wg, bg, gbias,
                                       qh, kh, vt, gb);
  attn_kernel<<<256, 256, 0, stream>>>(bias, qh, kh, vt, oh);
  outproj_kernel<<<128, 256, 0, stream>>>(oh, Wout, bout, gb, out);
}
